// Mamba_56805237457494
// MI455X (gfx1250) — matrix operand order repaired and checked by reading, not compiled, not run on hardware
//
#include <hip/hip_runtime.h>

// ---------------- model constants ----------------
constexpr int kNB = 4;            // mamba blocks
constexpr int kB  = 4;            // batch
constexpr int kL  = 2048;         // seqlen
constexpr int kIn = 64;           // input dim
constexpr int kOut= 128;          // output dim
constexpr int kH  = 256;          // hidden
constexpr int kDI = 512;          // d_inner
constexpr int kDS = 16;           // d_state
constexpr int kDR = 16;           // dt_rank
constexpr int kDRp= 32;           // dt_rank zero-padded to K-step
constexpr int kDX = 64;           // xdbl row stride (48 padded to 64 for guard-free GEMM)
constexpr int kNC = 32;           // scan chunks along L
constexpr int kCL = kL / kNC;     // 64 steps per chunk
constexpr int kM  = kB * kL;      // 8192 GEMM rows

typedef __attribute__((ext_vector_type(16))) __bf16          v16bf;
typedef __attribute__((ext_vector_type(8)))  float           v8f;
typedef __attribute__((ext_vector_type(8)))  unsigned short  ush8;

__device__ __forceinline__ unsigned short f2bf(float f) {
  unsigned int u = __float_as_uint(f);
  u += 0x7fffu + ((u >> 16) & 1u);          // round-to-nearest-even
  return (unsigned short)(u >> 16);
}
__device__ __forceinline__ float sigm(float x) { return 1.0f / (1.0f + __expf(-x)); }

// ---------------- prep: fp32 -> bf16 (flat) ----------------
__global__ void k_cvt_bf16(const float* __restrict__ s, unsigned short* __restrict__ d, int n) {
  int i = blockIdx.x * blockDim.x + threadIdx.x;
  if (i < n) d[i] = f2bf(s[i]);
}

// fp32 [K,N] -> bf16 [Np,Kp] transposed, zero-padded in both dims (guard-free GEMM loads)
__global__ void k_transpose_bf16(const float* __restrict__ s, unsigned short* __restrict__ d,
                                 int K, int N, int Kp, int Np) {
  int i = blockIdx.x * blockDim.x + threadIdx.x;
  if (i >= Np * Kp) return;
  int n = i / Kp, k = i % Kp;
  d[i] = (k < K && n < N) ? f2bf(s[(size_t)k * N + n]) : (unsigned short)0;
}

// ---------------- WMMA GEMM: C[M,N] = A_bf16[M,K] @ Wt_bf16[N,K]^T (+bias) ----------------
// One wave per 16x64 strip (4 accumulators, A fragment reused across 4 WMMAs).
// Preconditions enforced by the host: M%16==0, N%64==0, K%32==0, grid covers work exactly
// -> NO control flow except the uniform K loop; EXEC is all-ones at every WMMA.
__global__ void k_gemm_bf16(const unsigned short* __restrict__ A,
                            const unsigned short* __restrict__ Wt,
                            const float* __restrict__ bias,
                            float* __restrict__ C, int N, int K) {
  int strips = N >> 6;                        // 64-wide strips along N
  int sid = blockIdx.x * (blockDim.x >> 5) + (threadIdx.x >> 5);
  int tm = sid / strips, ts = sid % strips;
  int kh  = threadIdx.x >> 4 & 1;             // lane half (ISA fragment layout)
  int l15 = threadIdx.x & 15;
  int n0  = ts * 64;

  // A fragment: lanes 0-15 K = e | 16+e, lanes 16-31 shifted by 8 -> two contiguous 16B runs
  const unsigned short* ap  = A  + (size_t)(tm * 16 + l15) * K + 8 * kh;
  // B fragment: same K order as A (lanes 0-15 K = e | 16+e, lanes 16-31 shifted by 8) -> two 16B runs
  const unsigned short* bp0 = Wt + (size_t)(n0 +  0 + l15) * K + 8 * kh;
  const unsigned short* bp1 = Wt + (size_t)(n0 + 16 + l15) * K + 8 * kh;
  const unsigned short* bp2 = Wt + (size_t)(n0 + 32 + l15) * K + 8 * kh;
  const unsigned short* bp3 = Wt + (size_t)(n0 + 48 + l15) * K + 8 * kh;

  v8f acc0 = {0.f,0.f,0.f,0.f,0.f,0.f,0.f,0.f};
  v8f acc1 = acc0, acc2 = acc0, acc3 = acc0;
  union U { v16bf v; ush8 h[2]; };

  for (int kk = 0; kk < K; kk += 32) {
    U a, b0, b1, b2, b3;
    a.h[0]  = *(const ush8*)(ap);
    a.h[1]  = *(const ush8*)(ap + 16);   ap  += 32;
    b0.h[0] = *(const ush8*)(bp0);
    b0.h[1] = *(const ush8*)(bp0 + 16);  bp0 += 32;
    b1.h[0] = *(const ush8*)(bp1);
    b1.h[1] = *(const ush8*)(bp1 + 16);  bp1 += 32;
    b2.h[0] = *(const ush8*)(bp2);
    b2.h[1] = *(const ush8*)(bp2 + 16);  bp2 += 32;
    b3.h[0] = *(const ush8*)(bp3);
    b3.h[1] = *(const ush8*)(bp3 + 16);  bp3 += 32;
    acc0 = __builtin_amdgcn_wmma_f32_16x16x32_bf16(false, a.v, false, b0.v, (short)0, acc0, false, false);
    acc1 = __builtin_amdgcn_wmma_f32_16x16x32_bf16(false, a.v, false, b1.v, (short)0, acc1, false, false);
    acc2 = __builtin_amdgcn_wmma_f32_16x16x32_bf16(false, a.v, false, b2.v, (short)0, acc2, false, false);
    acc3 = __builtin_amdgcn_wmma_f32_16x16x32_bf16(false, a.v, false, b3.v, (short)0, acc3, false, false);
  }

  // C layout: VGPR v -> row tm*16 + v + 8*kh, column n0 + j*16 + l15
  int col = n0 + l15;
  float bv0 = bias ? bias[col     ] : 0.0f;
  float bv1 = bias ? bias[col + 16] : 0.0f;
  float bv2 = bias ? bias[col + 32] : 0.0f;
  float bv3 = bias ? bias[col + 48] : 0.0f;
  float* crow = C + (size_t)(tm * 16 + 8 * kh) * N + col;
  #pragma unroll
  for (int v = 0; v < 8; ++v) {
    crow[0]  = acc0[v] + bv0;
    crow[16] = acc1[v] + bv1;
    crow[32] = acc2[v] + bv2;
    crow[48] = acc3[v] + bv3;
    crow += N;
  }
}

// ---------------- layernorm (wave per row) -> bf16 ----------------
__global__ void k_layernorm(const float* __restrict__ h, const float* __restrict__ g,
                            const float* __restrict__ bt, unsigned short* __restrict__ out) {
  int wave = threadIdx.x >> 5, lane = threadIdx.x & 31;
  int row = blockIdx.x * (blockDim.x >> 5) + wave;
  if (row >= kM) return;
  const float* hr = h + (size_t)row * kH;
  float x[8], s = 0.f, sq = 0.f;
  #pragma unroll
  for (int e = 0; e < 8; ++e) { x[e] = hr[lane + 32 * e]; s += x[e]; sq += x[e] * x[e]; }
  #pragma unroll
  for (int m = 16; m >= 1; m >>= 1) { s += __shfl_xor(s, m, 32); sq += __shfl_xor(sq, m, 32); }
  float mean = s * (1.0f / kH);
  float var  = sq * (1.0f / kH) - mean * mean;
  float rstd = rsqrtf(var + 1e-5f);
  unsigned short* orow = out + (size_t)row * kH;
  #pragma unroll
  for (int e = 0; e < 8; ++e) {
    int j = lane + 32 * e;
    orow[j] = f2bf((x[e] - mean) * rstd * g[j] + bt[j]);
  }
}

// ---------------- causal depthwise conv (k=4) + SiLU; xi lives in xz[:, :kDI] ----------------
__global__ void k_conv_silu(const float* __restrict__ xz, const float* __restrict__ cw,
                            const float* __restrict__ cb, float* __restrict__ xc,
                            unsigned short* __restrict__ xcb) {
  int i = blockIdx.x * blockDim.x + threadIdx.x;
  if (i >= kB * kL * kDI) return;
  int d = i % kDI;
  int t = (i / kDI) % kL;
  int b = i / (kDI * kL);
  float acc = cb[d];
  #pragma unroll
  for (int k = 0; k < 4; ++k) {
    int tt = t + k - 3;
    if (tt >= 0) acc += cw[d * 4 + k] * xz[((size_t)(b * kL + tt)) * (2 * kDI) + d];
  }
  float v = acc * sigm(acc);
  xc[i] = v;
  xcb[i] = f2bf(v);
}

// xdbl[:, :16] -> bf16 dt-projection input, zero-padded to K=32
__global__ void k_dtin(const float* __restrict__ xdbl, unsigned short* __restrict__ d) {
  int i = blockIdx.x * blockDim.x + threadIdx.x;
  if (i >= kM * kDRp) return;
  int row = i / kDRp, j = i % kDRp;
  d[i] = (j < kDR) ? f2bf(xdbl[(size_t)row * kDX + j]) : (unsigned short)0;
}

__global__ void k_softplus(float* __restrict__ dt, int n) {
  int i = blockIdx.x * blockDim.x + threadIdx.x;
  if (i >= n) return;
  float x = dt[i];
  dt[i] = (x > 20.0f) ? x : log1pf(__expf(x));
}

// ---------------- chunked selective scan ----------------
// pass 1: chunk-local scan (h_in = 0), per-chunk decay product P and final state Hc
__global__ void k_scan1(const float* __restrict__ dt, const float* __restrict__ xc,
                        const float* __restrict__ xdbl, const float* __restrict__ Alog,
                        float* __restrict__ ysc, float* __restrict__ P, float* __restrict__ Hc) {
  int tid = blockIdx.x * blockDim.x + threadIdx.x;
  if (tid >= kB * kDI * kNC) return;
  int c = tid % kNC, d = (tid / kNC) % kDI, b = tid / (kNC * kDI);
  float a[kDS], h[kDS], p[kDS];
  #pragma unroll
  for (int s = 0; s < kDS; ++s) { a[s] = -__expf(Alog[d * kDS + s]); h[s] = 0.f; p[s] = 1.f; }
  for (int t = c * kCL; t < (c + 1) * kCL; ++t) {
    size_t r = (size_t)(b * kL + t);
    float dtv = dt[r * kDI + d];
    float xcv = xc[r * kDI + d];
    const float* xr = xdbl + r * kDX;
    float acc = 0.f;
    #pragma unroll
    for (int s = 0; s < kDS; ++s) {
      float w = __expf(dtv * a[s]);
      h[s] = w * h[s] + dtv * xr[16 + s] * xcv;   // dA*h + dt*B*x
      p[s] *= w;
      acc += h[s] * xr[32 + s];                   // y += h . C
    }
    ysc[r * kDI + d] = acc;
  }
  size_t o = ((size_t)((b * kDI + d) * kNC + c)) * kDS;
  #pragma unroll
  for (int s = 0; s < kDS; ++s) { P[o + s] = p[s]; Hc[o + s] = h[s]; }
}

// tiny serial stitch across the 32 chunks (per (b,d) channel)
__global__ void k_stitch(const float* __restrict__ P, const float* __restrict__ Hc,
                         float* __restrict__ hin) {
  int tid = blockIdx.x * blockDim.x + threadIdx.x;
  if (tid >= kB * kDI) return;
  float h[kDS];
  #pragma unroll
  for (int s = 0; s < kDS; ++s) h[s] = 0.f;
  for (int c = 0; c < kNC; ++c) {
    size_t o = ((size_t)(tid * kNC + c)) * kDS;
    #pragma unroll
    for (int s = 0; s < kDS; ++s) { hin[o + s] = h[s]; h[s] = P[o + s] * h[s] + Hc[o + s]; }
  }
}

// pass 2: add carried-state contribution, fuse D-skip + SiLU(z) gate, emit bf16 for out_W GEMM
__global__ void k_scan2(const float* __restrict__ dt, const float* __restrict__ xc,
                        const float* __restrict__ xdbl, const float* __restrict__ Alog,
                        const float* __restrict__ hin, const float* __restrict__ ysc,
                        const float* __restrict__ Dsk, const float* __restrict__ xz,
                        unsigned short* __restrict__ ybf) {
  int tid = blockIdx.x * blockDim.x + threadIdx.x;
  if (tid >= kB * kDI * kNC) return;
  int c = tid % kNC, d = (tid / kNC) % kDI, b = tid / (kNC * kDI);
  float a[kDS], pr[kDS], hi[kDS];
  size_t o = ((size_t)((b * kDI + d) * kNC + c)) * kDS;
  #pragma unroll
  for (int s = 0; s < kDS; ++s) { a[s] = -__expf(Alog[d * kDS + s]); pr[s] = 1.f; hi[s] = hin[o + s]; }
  float dsk = Dsk[d];
  for (int t = c * kCL; t < (c + 1) * kCL; ++t) {
    size_t r = (size_t)(b * kL + t);
    float dtv = dt[r * kDI + d];
    const float* xr = xdbl + r * kDX;
    float add = 0.f;
    #pragma unroll
    for (int s = 0; s < kDS; ++s) {
      pr[s] *= __expf(dtv * a[s]);                // cumulative decay incl. step t
      add += pr[s] * hi[s] * xr[32 + s];
    }
    float y = ysc[r * kDI + d] + add + dsk * xc[r * kDI + d];
    float z = xz[r * (2 * kDI) + kDI + d];
    y *= z * sigm(z);
    ybf[r * kDI + d] = f2bf(y);
  }
}

// gelu (tanh approx, as jax.nn.gelu) -> bf16
__global__ void k_gelu_bf16(const float* __restrict__ s, unsigned short* __restrict__ d, int n) {
  int i = blockIdx.x * blockDim.x + threadIdx.x;
  if (i >= n) return;
  float x = s[i];
  float t = 0.7978845608028654f * (x + 0.044715f * x * x * x);
  d[i] = f2bf(0.5f * x * (1.0f + tanhf(t)));
}

// GLU gate + residual add (in-place on h)
__global__ void k_glu_resid(const float* __restrict__ g, float* __restrict__ h) {
  int i = blockIdx.x * blockDim.x + threadIdx.x;
  if (i >= kM * kH) return;
  int row = i / kH, j = i % kH;
  float a  = g[(size_t)row * (2 * kH) + j];
  float bg = g[(size_t)row * (2 * kH) + kH + j];
  h[i] += a * sigm(bg);
}

__global__ void k_meanpool(const float* __restrict__ h, float* __restrict__ hm) {
  int i = blockIdx.x * blockDim.x + threadIdx.x;
  if (i >= kB * kH) return;
  int b = i / kH, j = i % kH;
  float s = 0.f;
  for (int t = 0; t < kL; ++t) s += h[((size_t)(b * kL + t)) * kH + j];
  hm[i] = s * (1.0f / kL);
}

// decoder matmul + softmax: one block per batch row, 128 threads
__global__ void k_dec_softmax(const float* __restrict__ hm, const float* __restrict__ W,
                              const float* __restrict__ bias, float* __restrict__ out) {
  __shared__ float red[kOut];
  int b = blockIdx.x, n = threadIdx.x;
  float logit = bias[n];
  for (int k = 0; k < kH; ++k) logit += hm[b * kH + k] * W[k * kOut + n];
  red[n] = logit; __syncthreads();
  for (int s = kOut / 2; s > 0; s >>= 1) { if (n < s) red[n] = fmaxf(red[n], red[n + s]); __syncthreads(); }
  float mx = red[0]; __syncthreads();
  float e = __expf(logit - mx);
  red[n] = e; __syncthreads();
  for (int s = kOut / 2; s > 0; s >>= 1) { if (n < s) red[n] += red[n + s]; __syncthreads(); }
  out[b * kOut + n] = e / red[0];
}

// ---------------- host orchestration ----------------
extern "C" void kernel_launch(void* const* d_in, const int* in_sizes, int n_in,
                              void* d_out, int out_size, void* d_ws, size_t ws_size,
                              hipStream_t stream) {
  const float* x       = (const float*)d_in[0];
  const float* enc_W   = (const float*)d_in[1];
  const float* enc_b   = (const float*)d_in[2];
  const float* ln_g    = (const float*)d_in[3];
  const float* ln_b    = (const float*)d_in[4];
  const float* in_W    = (const float*)d_in[5];
  const float* conv_W  = (const float*)d_in[6];
  const float* conv_b  = (const float*)d_in[7];
  const float* xproj_W = (const float*)d_in[8];
  const float* dtproj_W= (const float*)d_in[9];
  const float* dtproj_b= (const float*)d_in[10];
  const float* A_log   = (const float*)d_in[11];
  const float* D_skip  = (const float*)d_in[12];
  const float* out_W   = (const float*)d_in[13];
  const float* glu_W   = (const float*)d_in[14];
  const float* glu_b   = (const float*)d_in[15];
  const float* dec_W   = (const float*)d_in[16];
  const float* dec_b   = (const float*)d_in[17];
  float* out = (float*)d_out;

  // workspace carve-up (~160 MB); 256B alignment keeps all b128 loads aligned
  size_t off = 0;
  auto alloc = [&](size_t bytes) -> char* {
    off = (off + 255) & ~(size_t)255;
    char* p = (char*)d_ws + off; off += bytes; return p;
  };
  unsigned short* wt_enc  = (unsigned short*)alloc((size_t)kH * kIn * 2);
  unsigned short* wt_in   = (unsigned short*)alloc((size_t)kNB * 2 * kDI * kH * 2);
  unsigned short* wt_xp   = (unsigned short*)alloc((size_t)kNB * kDX * kDI * 2);
  unsigned short* wt_dt   = (unsigned short*)alloc((size_t)kNB * kDI * kDRp * 2);
  unsigned short* wt_out  = (unsigned short*)alloc((size_t)kNB * kH * kDI * 2);
  unsigned short* wt_glu  = (unsigned short*)alloc((size_t)kNB * 2 * kH * kH * 2);
  unsigned short* x_bf    = (unsigned short*)alloc((size_t)kM * kIn * 2);
  float*          hbuf    = (float*)alloc((size_t)kM * kH * 4);
  unsigned short* u_bf    = (unsigned short*)alloc((size_t)kM * kH * 2);
  float*          xz      = (float*)alloc((size_t)kM * 2 * kDI * 4);
  float*          xc      = (float*)alloc((size_t)kM * kDI * 4);
  unsigned short* xc_bf   = (unsigned short*)alloc((size_t)kM * kDI * 2);
  float*          xdbl    = (float*)alloc((size_t)kM * kDX * 4);
  unsigned short* dtin    = (unsigned short*)alloc((size_t)kM * kDRp * 2);
  float*          dtb     = (float*)alloc((size_t)kM * kDI * 4);
  float*          ysc     = (float*)alloc((size_t)kM * kDI * 4);
  unsigned short* y_bf    = (unsigned short*)alloc((size_t)kM * kDI * 2);
  float*          mo      = (float*)alloc((size_t)kM * kH * 4);
  unsigned short* mo_bf   = (unsigned short*)alloc((size_t)kM * kH * 2);
  float*          gbuf    = (float*)alloc((size_t)kM * 2 * kH * 4);
  float*          Pw      = (float*)alloc((size_t)kB * kDI * kNC * kDS * 4);
  float*          Hw      = (float*)alloc((size_t)kB * kDI * kNC * kDS * 4);
  float*          hinw    = (float*)alloc((size_t)kB * kDI * kNC * kDS * 4);
  float*          hmean   = (float*)alloc((size_t)kB * kH * 4);
  (void)ws_size; (void)in_sizes; (void)n_in; (void)out_size;

  auto cdiv = [](int a, int b) { return (a + b - 1) / b; };
  // N%64==0, M=8192 -> strip count divisible by 8: exact grid, no bounds check in kernel
  auto gemm = [&](const unsigned short* A, const unsigned short* Wt, const float* bias,
                  float* C, int N, int K) {
    int waves = (kM >> 4) * (N >> 6);
    k_gemm_bf16<<<waves / 8, 256, 0, stream>>>(A, Wt, bias, C, N, K);
  };
  auto twt = [&](const float* src, unsigned short* dst, int K, int N, int Kp, int Np) {
    k_transpose_bf16<<<cdiv(Np * Kp, 256), 256, 0, stream>>>(src, dst, K, N, Kp, Np);
  };

  // ---- prep: weight transpose+bf16 (+K/N padding), input bf16 ----
  k_cvt_bf16<<<cdiv(kM * kIn, 256), 256, 0, stream>>>(x, x_bf, kM * kIn);
  twt(enc_W, wt_enc, kIn, kH, kIn, kH);
  for (int i = 0; i < kNB; ++i) {
    twt(in_W    + (size_t)i * kH  * 2 * kDI, wt_in  + (size_t)i * 2 * kDI * kH, kH,  2 * kDI, kH,   2 * kDI);
    twt(xproj_W + (size_t)i * kDI * 48,      wt_xp  + (size_t)i * kDX * kDI,    kDI, 48,      kDI,  kDX);
    twt(dtproj_W+ (size_t)i * kDR * kDI,     wt_dt  + (size_t)i * kDI * kDRp,   kDR, kDI,     kDRp, kDI);
    twt(out_W   + (size_t)i * kDI * kH,      wt_out + (size_t)i * kH * kDI,     kDI, kH,      kDI,  kH);
    twt(glu_W   + (size_t)i * kH * 2 * kH,   wt_glu + (size_t)i * 2 * kH * kH,  kH,  2 * kH,  kH,   2 * kH);
  }

  // ---- encoder ----
  gemm(x_bf, wt_enc, enc_b, hbuf, kH, kIn);

  // ---- mamba blocks ----
  for (int i = 0; i < kNB; ++i) {
    k_layernorm<<<kM / 8, 256, 0, stream>>>(hbuf, ln_g + i * kH, ln_b + i * kH, u_bf);
    gemm(u_bf, wt_in + (size_t)i * 2 * kDI * kH, nullptr, xz, 2 * kDI, kH);
    k_conv_silu<<<cdiv(kM * kDI, 256), 256, 0, stream>>>(
        xz, conv_W + (size_t)i * kDI * 4, conv_b + i * kDI, xc, xc_bf);
    gemm(xc_bf, wt_xp + (size_t)i * kDX * kDI, nullptr, xdbl, kDX, kDI);
    k_dtin<<<cdiv(kM * kDRp, 256), 256, 0, stream>>>(xdbl, dtin);
    gemm(dtin, wt_dt + (size_t)i * kDI * kDRp, dtproj_b + i * kDI, dtb, kDI, kDRp);
    k_softplus<<<cdiv(kM * kDI, 256), 256, 0, stream>>>(dtb, kM * kDI);
    const float* Al = A_log + (size_t)i * kDI * kDS;
    k_scan1<<<cdiv(kB * kDI * kNC, 256), 256, 0, stream>>>(dtb, xc, xdbl, Al, ysc, Pw, Hw);
    k_stitch<<<cdiv(kB * kDI, 256), 256, 0, stream>>>(Pw, Hw, hinw);
    k_scan2<<<cdiv(kB * kDI * kNC, 256), 256, 0, stream>>>(
        dtb, xc, xdbl, Al, hinw, ysc, D_skip + i * kDI, xz, y_bf);
    gemm(y_bf, wt_out + (size_t)i * kH * kDI, nullptr, mo, kH, kDI);
    k_gelu_bf16<<<cdiv(kM * kH, 256), 256, 0, stream>>>(mo, mo_bf, kM * kH);
    gemm(mo_bf, wt_glu + (size_t)i * 2 * kH * kH, glu_b + i * 2 * kH, gbuf, 2 * kH, kH);
    k_glu_resid<<<cdiv(kM * kH, 256), 256, 0, stream>>>(gbuf, hbuf);
  }

  // ---- pool + decoder + softmax ----
  k_meanpool<<<cdiv(kB * kH, 256), 256, 0, stream>>>(hbuf, hmean);
  k_dec_softmax<<<kB, kOut, 0, stream>>>(hmean, dec_W, dec_b, out);
}
